// SlidingWindowAttention_70128226009192
// MI455X (gfx1250) — matrix operand order repaired and checked by reading, not compiled, not run on hardware
//
#include <hip/hip_runtime.h>
#include <hip/hip_bf16.h>

// Sliding-window attention, bf16 WMMA pipeline for gfx1250 (MI455X).
// Workspace layout (bytes, 256-aligned each):
//   xb    : x cast to bf16                [8192][768]        12.6 MB
//   wab   : W_attn^T bf16                 [2304][768]         3.5 MB
//   wpb   : W_proj^T bf16                 [768][768]          1.2 MB
//   qkb   : Q,K bf16                      [8192][1536]       25.2 MB
//   vtb   : V transposed bf16             [2][6][128][4096]  12.6 MB
//   ab    : attention output bf16         [8192][768]        12.6 MB
// total ~68 MB (fits in the 192 MB L2 -> inter-kernel traffic stays on chip)

#define SEQ     4096
#define HID     768
#define NH      6
#define HD      128
#define WIN     1024
#define MROWS   8192           // B*S
#define QKLD    1536           // Q+K columns
#define SCALE_F 0.08838834764831845f

typedef __attribute__((ext_vector_type(16))) __bf16 v16bf;
typedef __attribute__((ext_vector_type(8)))  __bf16 bf16x8;
typedef __attribute__((ext_vector_type(8)))  float  v8f;

static __device__ __forceinline__ v16bf ldpair(const __bf16* p0, const __bf16* p1) {
    bf16x8 lo = *reinterpret_cast<const bf16x8*>(p0);   // global_load_b128
    bf16x8 hi = *reinterpret_cast<const bf16x8*>(p1);   // global_load_b128
    v16bf r;
#pragma unroll
    for (int i = 0; i < 8; ++i) { r[i] = lo[i]; r[i + 8] = hi[i]; }
    return r;
}

static __device__ __forceinline__ v8f wmma_bf16(v16bf a, v16bf b, v8f c) {
    return __builtin_amdgcn_wmma_f32_16x16x32_bf16(false, a, false, b, (short)0, c,
                                                   false, false);
}

// ---------------- cast kernels ----------------

__global__ void k_cast_bf16(const float* __restrict__ in, __bf16* __restrict__ out, int n) {
    int i = blockIdx.x * blockDim.x + threadIdx.x;
    if (i < n) out[i] = (__bf16)in[i];
}

// in[K][N] (row-major) -> out[N][K] bf16  (coalesced writes)
__global__ void k_cast_transpose(const float* __restrict__ in, __bf16* __restrict__ out,
                                 int K, int N) {
    int i = blockIdx.x * blockDim.x + threadIdx.x;
    if (i < K * N) {
        int nn = i / K, kk = i - nn * K;
        out[i] = (__bf16)in[(size_t)kk * N + nn];
    }
}

// ---------------- QKV GEMM: qkv = xb @ W_attn ----------------
// Register-blocked: each wave computes a 32(M) x 64(N) tile (2x4 WMMA tiles,
// 8 f32 accumulators), reusing 2 A-fragments across 4 B-fragments per k-step:
// 12 lane-b128 loads per 8 WMMAs. Block (32,4) covers 64M x 128N.
// Q,K columns stored row-major into qkb; V columns stored transposed into vtb.

__global__ __launch_bounds__(128) void k_gemm_qkv(const __bf16* __restrict__ A,
                                                  const __bf16* __restrict__ Bt,
                                                  __bf16* __restrict__ qk,
                                                  __bf16* __restrict__ vt) {
    const int lane = threadIdx.x;
    const int h = lane >> 4, r = lane & 15;
    const int wm = threadIdx.y & 1, wn = threadIdx.y >> 1;
    const int m0 = blockIdx.y * 64 + wm * 32;
    const int n0 = blockIdx.x * 128 + wn * 64;

    const __bf16* Ar0 = A + (size_t)(m0 + r) * HID;
    const __bf16* Ar1 = A + (size_t)(m0 + 16 + r) * HID;
    const __bf16* Bc0 = Bt + (size_t)(n0 + r) * HID;

    v8f acc[2][4];
#pragma unroll
    for (int i = 0; i < 2; ++i)
#pragma unroll
        for (int j = 0; j < 4; ++j)
#pragma unroll
            for (int e = 0; e < 8; ++e) acc[i][j][e] = 0.0f;

#pragma unroll 2
    for (int k0 = 0; k0 < HID; k0 += 32) {
        v16bf a0 = ldpair(Ar0 + k0 + 8 * h, Ar0 + k0 + 16 + 8 * h);
        v16bf a1 = ldpair(Ar1 + k0 + 8 * h, Ar1 + k0 + 16 + 8 * h);
#pragma unroll
        for (int j = 0; j < 4; ++j) {
            const __bf16* Bc = Bc0 + (size_t)(j * 16) * HID;
            v16bf b = ldpair(Bc + k0 + 8 * h, Bc + k0 + 16 + 8 * h);
            acc[0][j] = wmma_bf16(a0, b, acc[0][j]);
            acc[1][j] = wmma_bf16(a1, b, acc[1][j]);
        }
    }

#pragma unroll
    for (int j = 0; j < 4; ++j) {
        const int n = n0 + j * 16 + r;
        if (n < QKLD) {                   // Q or K column: row-major store
#pragma unroll
            for (int i = 0; i < 2; ++i)
#pragma unroll
                for (int v = 0; v < 8; ++v) {
                    int m = m0 + i * 16 + v + 8 * h;
                    qk[(size_t)m * QKLD + n] = (__bf16)acc[i][j][v];
                }
        } else {                          // V column: store transposed [b][h][d][s]
            const int hh = (n - QKLD) >> 7, d = (n - QKLD) & 127;
#pragma unroll
            for (int i = 0; i < 2; ++i)
#pragma unroll
                for (int v = 0; v < 8; ++v) {
                    int m = m0 + i * 16 + v + 8 * h;
                    int bb = m >> 12, s = m & (SEQ - 1);
                    vt[((size_t)(bb * NH + hh) * HD + d) * SEQ + s] = (__bf16)acc[i][j][v];
                }
        }
    }
}

// ---------------- sliding-window flash attention ----------------
// block = (32,4): 4 waves, each wave owns a 16-query block of one (b,head).
// Q fragments stay register-resident across the whole 1024-key window.

__global__ __launch_bounds__(128) void k_attn(const __bf16* __restrict__ qk,
                                              const __bf16* __restrict__ vt,
                                              __bf16* __restrict__ attno) {
    __shared__ __bf16 Pl[4][16 * 32];     // per-wave P staging (C-layout -> A-layout)
    const int lane = threadIdx.x, w = threadIdx.y;
    const int h = lane >> 4, r = lane & 15;
    const int bz = blockIdx.z, hy = blockIdx.y;
    const int q0 = blockIdx.x * 64 + w * 16;

    const __bf16* Qb = qk + ((size_t)bz * SEQ + q0) * QKLD + hy * HD;
    const __bf16* Kb = qk + (size_t)bz * SEQ * QKLD + HID + hy * HD;
    const __bf16* Vh = vt + (size_t)(bz * NH + hy) * HD * SEQ;
    __bf16* Pw = Pl[w];

    // Preload Q fragments (16 x 128 over 4 K-steps of 32)
    v16bf qf[4];
#pragma unroll
    for (int t = 0; t < 4; ++t)
        qf[t] = ldpair(Qb + (size_t)r * QKLD + t * 32 + 8 * h,
                       Qb + (size_t)r * QKLD + t * 32 + 16 + 8 * h);

    float mrow[8], lrow[8];
    v8f acc[8];
#pragma unroll
    for (int v = 0; v < 8; ++v) { mrow[v] = -1e30f; lrow[v] = 0.0f; }
#pragma unroll
    for (int t = 0; t < 8; ++t)
#pragma unroll
        for (int v = 0; v < 8; ++v) acc[t][v] = 0.0f;

    int kstart = q0 - (WIN - 1);
    if (kstart < 0) kstart = 0;
    kstart &= ~31;

    for (int kb = kstart; kb <= q0 + 15; kb += 32) {
        // S tiles: keys [kb, kb+16) and [kb+16, kb+32)
        v8f s0, s1;
#pragma unroll
        for (int i = 0; i < 8; ++i) { s0[i] = 0.0f; s1[i] = 0.0f; }
        const __bf16* K0 = Kb + (size_t)(kb + r) * QKLD;
        const __bf16* K1 = Kb + (size_t)(kb + 16 + r) * QKLD;
#pragma unroll
        for (int t = 0; t < 4; ++t) {
            v16bf f0 = ldpair(K0 + t * 32 + 8 * h, K0 + t * 32 + 16 + 8 * h);
            v16bf f1 = ldpair(K1 + t * 32 + 8 * h, K1 + t * 32 + 16 + 8 * h);
            s0 = wmma_bf16(qf[t], f0, s0);
            s1 = wmma_bf16(qf[t], f1, s1);
        }
        // Masked online softmax (C layout: row m = v+8h, col = lane&15)
#pragma unroll
        for (int v = 0; v < 8; ++v) {
            const int m = v + 8 * h, q = q0 + m;
            const int key0 = kb + r, key1 = key0 + 16;
            const bool ok0 = (key0 <= q) && (q - key0 < WIN);
            const bool ok1 = (key1 <= q) && (q - key1 < WIN);
            float x0 = ok0 ? s0[v] * SCALE_F : -1e30f;
            float x1 = ok1 ? s1[v] * SCALE_F : -1e30f;
            float tmax = fmaxf(x0, x1);
#pragma unroll
            for (int o = 1; o < 16; o <<= 1) tmax = fmaxf(tmax, __shfl_xor(tmax, o, 32));
            const float nm   = fmaxf(mrow[v], tmax);
            const float corr = __expf(mrow[v] - nm);
            const float p0 = ok0 ? __expf(x0 - nm) : 0.0f;
            const float p1 = ok1 ? __expf(x1 - nm) : 0.0f;
            float ps = p0 + p1;
#pragma unroll
            for (int o = 1; o < 16; o <<= 1) ps += __shfl_xor(ps, o, 32);
            mrow[v] = nm;
            lrow[v] = lrow[v] * corr + ps;
#pragma unroll
            for (int t = 0; t < 8; ++t) acc[t][v] *= corr;
            Pw[m * 32 + r]      = (__bf16)p0;
            Pw[m * 32 + 16 + r] = (__bf16)p1;
        }
        // Re-read P in A layout (wave-local LDS; DS ops are in-order per wave)
        v16bf pf = ldpair(Pw + r * 32 + 8 * h, Pw + r * 32 + 16 + 8 * h);
        // O += P @ V ; V is transposed [d][s], so B frags are contiguous b128 loads
#pragma unroll
        for (int t = 0; t < 8; ++t) {
            const __bf16* Vp = Vh + (size_t)(t * 16 + r) * SEQ + kb + 8 * h;
            v16bf vf = ldpair(Vp, Vp + 16);
            acc[t] = wmma_bf16(pf, vf, acc[t]);
        }
    }

    // Normalize and store bf16 attention output [b][s][h*128 + d]
    float inv[8];
#pragma unroll
    for (int v = 0; v < 8; ++v) inv[v] = 1.0f / lrow[v];
    __bf16* Ob = attno + (size_t)bz * SEQ * HID + hy * HD;
#pragma unroll
    for (int t = 0; t < 8; ++t)
#pragma unroll
        for (int v = 0; v < 8; ++v) {
            int m = v + 8 * h;
            Ob[(size_t)(q0 + m) * HID + t * 16 + r] = (__bf16)(acc[t][v] * inv[v]);
        }
}

// ---------------- output projection: out = attno @ W_proj (fp32 store) ----------------
// Same 32x64 register blocking as the QKV GEMM; fp32 stores to d_out.

__global__ __launch_bounds__(128) void k_gemm_proj(const __bf16* __restrict__ A,
                                                   const __bf16* __restrict__ Bt,
                                                   float* __restrict__ out) {
    const int lane = threadIdx.x;
    const int h = lane >> 4, r = lane & 15;
    const int wm = threadIdx.y & 1, wn = threadIdx.y >> 1;
    const int m0 = blockIdx.y * 64 + wm * 32;
    const int n0 = blockIdx.x * 128 + wn * 64;

    const __bf16* Ar0 = A + (size_t)(m0 + r) * HID;
    const __bf16* Ar1 = A + (size_t)(m0 + 16 + r) * HID;
    const __bf16* Bc0 = Bt + (size_t)(n0 + r) * HID;

    v8f acc[2][4];
#pragma unroll
    for (int i = 0; i < 2; ++i)
#pragma unroll
        for (int j = 0; j < 4; ++j)
#pragma unroll
            for (int e = 0; e < 8; ++e) acc[i][j][e] = 0.0f;

#pragma unroll 2
    for (int k0 = 0; k0 < HID; k0 += 32) {
        v16bf a0 = ldpair(Ar0 + k0 + 8 * h, Ar0 + k0 + 16 + 8 * h);
        v16bf a1 = ldpair(Ar1 + k0 + 8 * h, Ar1 + k0 + 16 + 8 * h);
#pragma unroll
        for (int j = 0; j < 4; ++j) {
            const __bf16* Bc = Bc0 + (size_t)(j * 16) * HID;
            v16bf b = ldpair(Bc + k0 + 8 * h, Bc + k0 + 16 + 8 * h);
            acc[0][j] = wmma_bf16(a0, b, acc[0][j]);
            acc[1][j] = wmma_bf16(a1, b, acc[1][j]);
        }
    }

#pragma unroll
    for (int j = 0; j < 4; ++j) {
        const int n = n0 + j * 16 + r;
#pragma unroll
        for (int i = 0; i < 2; ++i)
#pragma unroll
            for (int v = 0; v < 8; ++v) {
                int m = m0 + i * 16 + v + 8 * h;
                out[(size_t)m * HID + n] = acc[i][j][v];
            }
    }
}

// ---------------- host launcher ----------------

extern "C" void kernel_launch(void* const* d_in, const int* in_sizes, int n_in,
                              void* d_out, int out_size, void* d_ws, size_t ws_size,
                              hipStream_t stream) {
    (void)in_sizes; (void)n_in; (void)out_size; (void)ws_size;
    const float* x  = (const float*)d_in[0];
    const float* Wa = (const float*)d_in[1];
    const float* Wp = (const float*)d_in[2];
    float* out = (float*)d_out;

    char* ws = (char*)d_ws;
    size_t off = 0;
    auto carve = [&](size_t bytes) -> void* {
        void* p = ws + off;
        off += (bytes + 255) & ~(size_t)255;
        return p;
    };
    __bf16* xb  = (__bf16*)carve((size_t)MROWS * HID * 2);
    __bf16* wab = (__bf16*)carve((size_t)(3 * HID) * HID * 2);
    __bf16* wpb = (__bf16*)carve((size_t)HID * HID * 2);
    __bf16* qkb = (__bf16*)carve((size_t)MROWS * QKLD * 2);
    __bf16* vtb = (__bf16*)carve((size_t)2 * NH * HD * SEQ * 2);
    __bf16* ab  = (__bf16*)carve((size_t)MROWS * HID * 2);

    const int nx = MROWS * HID;           // 6,291,456
    const int nwa = HID * 3 * HID;        // 1,769,472
    const int nwp = HID * HID;            //   589,824
    k_cast_bf16     <<<(nx  + 255) / 256, 256, 0, stream>>>(x,  xb,  nx);
    k_cast_transpose<<<(nwa + 255) / 256, 256, 0, stream>>>(Wa, wab, HID, 3 * HID);
    k_cast_transpose<<<(nwp + 255) / 256, 256, 0, stream>>>(Wp, wpb, HID, HID);

    k_gemm_qkv <<<dim3((3 * HID) / 128, MROWS / 64), dim3(32, 4), 0, stream>>>(xb, wab, qkb, vtb);
    k_attn     <<<dim3(SEQ / 64, NH, 2),             dim3(32, 4), 0, stream>>>(qkb, vtb, ab);
    k_gemm_proj<<<dim3(HID / 128, MROWS / 64),       dim3(32, 4), 0, stream>>>(ab, wpb, out);
}
